// HybridRecSystem_33629593927760
// MI455X (gfx1250) — matrix operand order repaired and checked by reading, not compiled, not run on hardware
//
#include <hip/hip_runtime.h>

#define BATCH   16384
#define EMB     64
#define UFEAT   21
#define IFEAT   39
#define HID     256
#define XU      (EMB + UFEAT)   /* 85  */
#define XI      (EMB + IFEAT)   /* 103 */
#define KU      96              /* 85 padded to 3*32  */
#define KI      128             /* 103 padded to 4*32 */
#define K1      128             /* 2*EMB, exact       */
#define TILES   (BATCH / 16)    /* 1024 */
#define WAVES   4
#define NTHR    (WAVES * 32)
#define NBLK    128

typedef __attribute__((ext_vector_type(16))) _Float16 v16h;
typedef __attribute__((ext_vector_type(8)))  _Float16 v8h;
typedef __attribute__((ext_vector_type(4)))  _Float16 v4h;
typedef __attribute__((ext_vector_type(8)))  float    v8f;
typedef __attribute__((ext_vector_type(4)))  float    v4f;

// Wave-internal LDS producer->consumer ordering (CDNA5 split counters).
__device__ __forceinline__ void wave_lds_fence() {
  asm volatile("s_wait_dscnt 0" ::: "memory");
}

// A fragment (16-bit A 16x32): lane M = lane&15, half = lane>>4.
// halves 0..7  <- K = 32s + 8*half + (0..7)      (contiguous, 16B)
// halves 8..15 <- K = 32s + 16 + 8*half + (0..7) (contiguous, 16B)
__device__ __forceinline__ v16h load_a_frag(const _Float16* p) {
  v8h lo = *(const v8h*)(p);
  v8h hi = *(const v8h*)(p + 16);
  v16h a;
#pragma unroll
  for (int i = 0; i < 8; ++i) { a[i] = lo[i]; a[i + 8] = hi[i]; }
  return a;
}

// B fragment (16-bit B 32x16): lane N = lane&15, halves 0..7 <- K = 8*half + (0..7), halves 8..15 <- K = 16 + 8*half + (0..7).
__device__ __forceinline__ v16h load_b_frag(const _Float16* p) {
  v8h lo = *(const v8h*)(p);
  v8h hi = *(const v8h*)(p + 16);
  v16h b;
#pragma unroll
  for (int i = 0; i < 8; ++i) { b[i] = lo[i]; b[i + 8] = hi[i]; }
  return b;
}

// Load all A fragments for one 16-row activation panel (K = 32*nk), reused
// across every N-tile of the layer.
__device__ __forceinline__ void load_a_frags(const _Float16* act_row, int half,
                                             int nk, v16h* afr) {
#pragma unroll
  for (int s = 0; s < 4; ++s)
    if (s < nk) afr[s] = load_a_frag(act_row + 32 * s + 8 * half);
}

// One 16x16 f32 output tile using pre-loaded A fragments.
__device__ __forceinline__ v8f mm_tile(const v16h* afr, const _Float16* w_row,
                                       int nk, int half) {
  v8f acc = {};
#pragma unroll
  for (int s = 0; s < 4; ++s) {
    if (s < nk) {
      v16h b = load_b_frag(w_row + 32 * s + 8 * half);
      acc = __builtin_amdgcn_wmma_f32_16x16x32_f16(false, afr[s], false, b,
                                                   (short)0, acc, false, false);
    }
  }
  return acc;
}

__global__ __launch_bounds__(NTHR)
void recsys_fused_kernel(const int* __restrict__ user_id, const int* __restrict__ item_id,
                         const float* __restrict__ ufeat, const float* __restrict__ ifeat,
                         const float* __restrict__ uemb,  const float* __restrict__ iemb,
                         const float* __restrict__ Wuf,   const float* __restrict__ buf_,
                         const float* __restrict__ Wif,   const float* __restrict__ bif_,
                         const float* __restrict__ W1,    const float* __restrict__ b1_,
                         const float* __restrict__ W2,    const float* __restrict__ b2_,
                         float* __restrict__ out) {
  // Weight images (f16, K zero-padded), shared by all waves: 12 + 16 + 64 KB.
  __shared__ alignas(16) _Float16 s_wuf[EMB * KU];
  __shared__ alignas(16) _Float16 s_wif[EMB * KI];
  __shared__ alignas(16) _Float16 s_w1[HID * K1];
  // Per-wave activation double-buffer + f32 residual buffer.
  __shared__ alignas(16) _Float16 s_act[WAVES][2][16][K1];
  __shared__ alignas(16) float    s_res[WAVES][16][EMB];
  __shared__ float s_bu[EMB], s_bi[EMB], s_b1[HID], s_w2[HID];

  const int tid = threadIdx.x;

  // ---- Stage weights f32 -> f16 LDS (once per block) ----
  for (int i = tid; i < EMB * KU; i += NTHR) {
    int r = i / KU, k = i - r * KU;
    s_wuf[i] = (k < XU) ? (_Float16)Wuf[r * XU + k] : (_Float16)0.0f;
  }
  for (int i = tid; i < EMB * KI; i += NTHR) {
    int r = i / KI, k = i - r * KI;
    s_wif[i] = (k < XI) ? (_Float16)Wif[r * XI + k] : (_Float16)0.0f;
  }
  for (int i = tid; i < HID * K1; i += NTHR) s_w1[i] = (_Float16)W1[i];
  for (int i = tid; i < EMB; i += NTHR) { s_bu[i] = buf_[i]; s_bi[i] = bif_[i]; }
  for (int i = tid; i < HID; i += NTHR) { s_b1[i] = b1_[i]; s_w2[i] = W2[i]; }
  __syncthreads();

  const int w    = tid >> 5;
  const int lane = tid & 31;
  const int nl   = lane & 15;   // A-row / B-col / C-col index
  const int half = lane >> 4;
  _Float16 (*actA)[K1] = s_act[w][0];
  _Float16 (*actB)[K1] = s_act[w][1];
  float    (*res)[EMB] = s_res[w];
  const float b2v = b2_[0];
  const int tstride = gridDim.x * WAVES;

  for (int t = blockIdx.x * WAVES + w; t < TILES; t += tstride) {
    const int row = t * 16 + nl;

    // Prefetch next tile's gathered embedding rows (global_prefetch_b8) so the
    // random HBM gather overlaps this tile's WMMA work.
    {
      const int tn = t + tstride;
      if (tn < TILES) {
        const int rown = tn * 16 + nl;
        __builtin_prefetch(uemb + (size_t)user_id[rown] * EMB, 0, 0);
        __builtin_prefetch(iemb + (size_t)item_id[rown] * EMB, 0, 0);
      }
    }

    // ---- x_u = relu([ue, uf]) -> actA (f16, pad to 96); res = ue (f32) ----
    {
      const float* uerow = uemb + (size_t)user_id[row] * EMB;
      const float* ufrow = ufeat + (size_t)row * UFEAT;
      const int k0 = half * 48, k1 = k0 + 48;
      // embedding span: 16B-aligned float4 path
      for (int k = k0; k + 3 < EMB && k < k1; k += 4) {
        v4f v = *(const v4f*)(uerow + k);
        *(v4f*)&res[nl][k] = v;
        v4h hv;
#pragma unroll
        for (int i = 0; i < 4; ++i) hv[i] = (_Float16)(v[i] > 0.f ? v[i] : 0.f);
        *(v4h*)&actA[nl][k] = hv;
      }
      // feature + zero-pad span (scalar; feature rows only 4B-aligned)
      for (int k = (k0 > EMB ? k0 : EMB); k < k1; ++k) {
        float x = 0.f;
        if (k < XU) { float v = ufrow[k - EMB]; x = v > 0.f ? v : 0.f; }
        actA[nl][k] = (_Float16)x;
      }
    }
    wave_lds_fence();

    // ---- uh = x_u @ Wuf^T + b_uf + ue ; actB[:,0:64] = relu(uh) ----
    {
      v16h afr[4];
      load_a_frags(&actA[nl][0], half, 3, afr);
#pragma unroll
      for (int nt = 0; nt < 4; ++nt) {
        const int n0 = nt * 16;
        v8f acc = mm_tile(afr, &s_wuf[(n0 + nl) * KU], 3, half);
        const float bb = s_bu[n0 + nl];
#pragma unroll
        for (int r = 0; r < 8; ++r) {
          const int mr = 8 * half + r;                    // C/D layout: M = 8*half + r
          float uh = acc[r] + bb + res[mr][n0 + nl];
          actB[mr][n0 + nl] = (_Float16)(uh > 0.f ? uh : 0.f);
        }
      }
    }
    wave_lds_fence();

    // ---- x_i = relu([ie, if]) -> actA (pad to 128); res = ie ----
    {
      const float* ierow = iemb + (size_t)item_id[row] * EMB;
      const float* ifrow = ifeat + (size_t)row * IFEAT;
      const int k0 = half * 64, k1 = k0 + 64;
      for (int k = k0; k + 3 < EMB && k < k1; k += 4) {
        v4f v = *(const v4f*)(ierow + k);
        *(v4f*)&res[nl][k] = v;
        v4h hv;
#pragma unroll
        for (int i = 0; i < 4; ++i) hv[i] = (_Float16)(v[i] > 0.f ? v[i] : 0.f);
        *(v4h*)&actA[nl][k] = hv;
      }
      for (int k = (k0 > EMB ? k0 : EMB); k < k1; ++k) {
        float x = 0.f;
        if (k < XI) { float v = ifrow[k - EMB]; x = v > 0.f ? v : 0.f; }
        actA[nl][k] = (_Float16)x;
      }
    }
    wave_lds_fence();

    // ---- ih = x_i @ Wif^T + b_if + ie ; actB[:,64:128] = relu(ih) ----
    {
      v16h afr[4];
      load_a_frags(&actA[nl][0], half, 4, afr);
#pragma unroll
      for (int nt = 0; nt < 4; ++nt) {
        const int n0 = nt * 16;
        v8f acc = mm_tile(afr, &s_wif[(n0 + nl) * KI], 4, half);
        const float bb = s_bi[n0 + nl];
#pragma unroll
        for (int r = 0; r < 8; ++r) {
          const int mr = 8 * half + r;
          float ih = acc[r] + bb + res[mr][n0 + nl];
          actB[mr][EMB + n0 + nl] = (_Float16)(ih > 0.f ? ih : 0.f);
        }
      }
    }
    wave_lds_fence();

    // ---- layer1 (K=128, N=256) fused with 256->1 head ----
    float oacc[8];
#pragma unroll
    for (int r = 0; r < 8; ++r) oacc[r] = 0.f;
    {
      v16h afr[4];
      load_a_frags(&actB[nl][0], half, 4, afr);           // reused for all 16 N-tiles
      for (int nt = 0; nt < 16; ++nt) {
        const int n0 = nt * 16;
        v8f acc = mm_tile(afr, &s_w1[(n0 + nl) * K1], 4, half);
        const float bb  = s_b1[n0 + nl];
        const float w2v = s_w2[n0 + nl];
#pragma unroll
        for (int r = 0; r < 8; ++r) {
          float h = acc[r] + bb;
          h = h > 0.f ? h : 0.f;
          oacc[r] += h * w2v;                             // head dot, per column
        }
      }
    }
    // Reduce the head dot across the 16 lanes of each half-wave.
#pragma unroll
    for (int r = 0; r < 8; ++r) {
      oacc[r] += __shfl_xor(oacc[r], 1, 16);
      oacc[r] += __shfl_xor(oacc[r], 2, 16);
      oacc[r] += __shfl_xor(oacc[r], 4, 16);
      oacc[r] += __shfl_xor(oacc[r], 8, 16);
    }
    if (nl == 0) {
#pragma unroll
      for (int r = 0; r < 8; ++r)
        out[t * 16 + 8 * half + r] = oacc[r] + b2v;
    }
    wave_lds_fence();   // before next iteration overwrites actA/res
  }
}

extern "C" void kernel_launch(void* const* d_in, const int* in_sizes, int n_in,
                              void* d_out, int out_size, void* d_ws, size_t ws_size,
                              hipStream_t stream) {
  (void)in_sizes; (void)n_in; (void)out_size; (void)d_ws; (void)ws_size;
  const int*   user_id = (const int*)d_in[0];
  const int*   item_id = (const int*)d_in[1];
  const float* ufeat   = (const float*)d_in[2];
  const float* ifeat   = (const float*)d_in[3];
  const float* uemb    = (const float*)d_in[4];
  const float* iemb    = (const float*)d_in[5];
  const float* Wuf     = (const float*)d_in[6];
  const float* b_uf    = (const float*)d_in[7];
  const float* Wif     = (const float*)d_in[8];
  const float* b_if    = (const float*)d_in[9];
  const float* W1      = (const float*)d_in[10];
  const float* b1      = (const float*)d_in[11];
  const float* W2      = (const float*)d_in[12];
  const float* b2      = (const float*)d_in[13];
  float* out = (float*)d_out;

  recsys_fused_kernel<<<dim3(NBLK), dim3(NTHR), 0, stream>>>(
      user_id, item_id, ufeat, ifeat, uemb, iemb,
      Wuf, b_uf, Wif, b_if, W1, b1, W2, b2, out);
}
